// DisentangledGraphConvEncoder_41059887350349
// MI455X (gfx1250) — matrix operand order repaired and checked by reading, not compiled, not run on hardware
//
#include <hip/hip_runtime.h>
#include <hip/hip_bf16.h>

typedef __attribute__((ext_vector_type(16))) _Float16 v16h;
typedef __attribute__((ext_vector_type(8)))  float    v8f;

#define NNODES 50000
#define NEDGES 800000
#define DIN    256
#define NCH    8
#define NH     64
#define CHW    (NCH * NH)   /* 512 */
#define LN_EPS 1e-5f

union H16x16 { v16h v; uint4 q[2]; };

// ---------------- precision / layout conversion kernels ----------------

__global__ void k_cvt_x(const float* __restrict__ x, _Float16* __restrict__ xh, int n) {
  int i = blockIdx.x * blockDim.x + threadIdx.x;
  if (i < n) xh[i] = (_Float16)x[i];
}

// proj [D][C*H] row-major -> projT [C*H][D] f16 (so WMMA B columns are contiguous)
__global__ void k_cvt_projT(const float* __restrict__ proj, _Float16* __restrict__ projT) {
  int i = blockIdx.x * blockDim.x + threadIdx.x;
  if (i >= DIN * CHW) return;
  int d = i / CHW, o = i % CHW;
  projT[o * DIN + d] = (_Float16)proj[i];
}

// W [C][H][H] -> WT [C][O][H] f16 (transpose within each channel)
__global__ void k_cvt_WT(const float* __restrict__ W, _Float16* __restrict__ WT) {
  int i = blockIdx.x * blockDim.x + threadIdx.x;
  if (i >= NCH * NH * NH) return;
  int c = i / (NH * NH);
  int r = i % (NH * NH);
  int h = r / NH, o = r % NH;
  WT[c * NH * NH + o * NH + h] = (_Float16)W[i];
}

// ---------------- projection GEMM: out[N,512] = xh[N,256] @ projT^T -----
// One wave computes a 16x64 output tile (4 accumulators), K=256 in 8 steps.
// Block = 8 waves sharing the same 16 rows (n0 = wave*64).

__global__ void __launch_bounds__(256)
k_proj_gemm(const _Float16* __restrict__ xh, const _Float16* __restrict__ projT,
            float* __restrict__ out) {
  const int lane = threadIdx.x & 31;
  const int wave = threadIdx.x >> 5;
  const int m0   = blockIdx.x * 16;
  const int n0   = wave * 64;
  const int col  = lane & 15;
  const int half = lane >> 4;

  const _Float16* arow = xh + (size_t)(m0 + col) * DIN;
  v8f acc0 = {}, acc1 = {}, acc2 = {}, acc3 = {};

  for (int k0 = 0; k0 < DIN; k0 += 32) {
    // A fragment: lane holds row m0+col, K = {k0+half*8..+7, k0+16+half*8..+7}
    H16x16 a;
    a.q[0] = *(const uint4*)(arow + k0 + half * 8);
    a.q[1] = *(const uint4*)(arow + k0 + 16 + half * 8);

    // B fragments: lane holds column n, K = {k0+half*8..+7, k0+16+half*8..+7}
    const _Float16* bb = projT + (size_t)(n0 + col) * DIN + k0 + half * 8;
    H16x16 b0, b1, b2, b3;
    b0.q[0] = *(const uint4*)(bb + 0 * 16 * DIN); b0.q[1] = *(const uint4*)(bb + 0 * 16 * DIN + 16);
    b1.q[0] = *(const uint4*)(bb + 1 * 16 * DIN); b1.q[1] = *(const uint4*)(bb + 1 * 16 * DIN + 16);
    b2.q[0] = *(const uint4*)(bb + 2 * 16 * DIN); b2.q[1] = *(const uint4*)(bb + 2 * 16 * DIN + 16);
    b3.q[0] = *(const uint4*)(bb + 3 * 16 * DIN); b3.q[1] = *(const uint4*)(bb + 3 * 16 * DIN + 16);

    acc0 = __builtin_amdgcn_wmma_f32_16x16x32_f16(false, a.v, false, b0.v, (short)0, acc0, false, false);
    acc1 = __builtin_amdgcn_wmma_f32_16x16x32_f16(false, a.v, false, b1.v, (short)0, acc1, false, false);
    acc2 = __builtin_amdgcn_wmma_f32_16x16x32_f16(false, a.v, false, b2.v, (short)0, acc2, false, false);
    acc3 = __builtin_amdgcn_wmma_f32_16x16x32_f16(false, a.v, false, b3.v, (short)0, acc3, false, false);
  }

  // C/D layout: VGPR r holds row m0 + half*8 + r, column n0 + t*16 + col
  const int rbase = m0 + half * 8;
#pragma unroll
  for (int r = 0; r < 8; ++r) {
    float* op = out + (size_t)(rbase + r) * CHW + n0 + col;
    op[0]  = acc0[r];
    op[16] = acc1[r];
    op[32] = acc2[r];
    op[48] = acc3[r];
  }
}

// ---------------- edge scatter: acc[dst] += omega[e,c] * h[src] ---------
// One wave per edge: 32 lanes x 16 contiguous floats = 512 values.
// Lane l covers channel c = l/4, so one omega read per lane.

__global__ void __launch_bounds__(256)
k_scatter(const float* __restrict__ h, const long long* __restrict__ ei,
          const float* __restrict__ omega, float* __restrict__ acc) {
  int gid = blockIdx.x * blockDim.x + threadIdx.x;
  int e = gid >> 5;
  if (e >= NEDGES) return;
  int lane = gid & 31;
  int src = (int)ei[e];
  int dst = (int)ei[NEDGES + e];
  int c = lane >> 2;
  float w = omega[(size_t)e * NCH + c];
  const float* hp = h + (size_t)src * CHW + lane * 16;
  float* ap = acc + (size_t)dst * CHW + lane * 16;
#pragma unroll
  for (int i = 0; i < 16; i += 4) {
    float4 v = *(const float4*)(hp + i);
    atomicAdd(ap + i + 0, v.x * w);
    atomicAdd(ap + i + 1, v.y * w);
    atomicAdd(ap + i + 2, v.z * w);
    atomicAdd(ap + i + 3, v.w * w);
  }
}

// ---------------- channel BMM: out[n,c,:] = acc[n,c,:] @ W[c] -----------
// One wave per (16-node tile, channel). A built on the fly f32->f16,
// B from pre-transposed f16 weights. 8 WMMAs per wave (K=64, Nout=64).

__global__ void __launch_bounds__(256)
k_bmm(const float* __restrict__ accb, const _Float16* __restrict__ WT,
      float* __restrict__ out) {
  const int lane = threadIdx.x & 31;
  const int c    = threadIdx.x >> 5;
  const int m0   = blockIdx.x * 16;
  const int col  = lane & 15;
  const int half = lane >> 4;

  const float*    arow = accb + (size_t)(m0 + col) * CHW + c * NH;
  const _Float16* wc   = WT + c * NH * NH;

  v8f acc0 = {}, acc1 = {}, acc2 = {}, acc3 = {};
#pragma unroll
  for (int k0 = 0; k0 < NH; k0 += 32) {
    const float* ap = arow + k0 + half * 8;
    float4 f0 = *(const float4*)(ap);
    float4 f1 = *(const float4*)(ap + 4);
    float4 f2 = *(const float4*)(ap + 16);
    float4 f3 = *(const float4*)(ap + 20);
    v16h a;
    a[0]  = (_Float16)f0.x; a[1]  = (_Float16)f0.y; a[2]  = (_Float16)f0.z; a[3]  = (_Float16)f0.w;
    a[4]  = (_Float16)f1.x; a[5]  = (_Float16)f1.y; a[6]  = (_Float16)f1.z; a[7]  = (_Float16)f1.w;
    a[8]  = (_Float16)f2.x; a[9]  = (_Float16)f2.y; a[10] = (_Float16)f2.z; a[11] = (_Float16)f2.w;
    a[12] = (_Float16)f3.x; a[13] = (_Float16)f3.y; a[14] = (_Float16)f3.z; a[15] = (_Float16)f3.w;

    const _Float16* bb = wc + (size_t)col * NH + k0 + half * 8;
    H16x16 b0, b1, b2, b3;
    b0.q[0] = *(const uint4*)(bb + 0 * 16 * NH); b0.q[1] = *(const uint4*)(bb + 0 * 16 * NH + 16);
    b1.q[0] = *(const uint4*)(bb + 1 * 16 * NH); b1.q[1] = *(const uint4*)(bb + 1 * 16 * NH + 16);
    b2.q[0] = *(const uint4*)(bb + 2 * 16 * NH); b2.q[1] = *(const uint4*)(bb + 2 * 16 * NH + 16);
    b3.q[0] = *(const uint4*)(bb + 3 * 16 * NH); b3.q[1] = *(const uint4*)(bb + 3 * 16 * NH + 16);

    acc0 = __builtin_amdgcn_wmma_f32_16x16x32_f16(false, a, false, b0.v, (short)0, acc0, false, false);
    acc1 = __builtin_amdgcn_wmma_f32_16x16x32_f16(false, a, false, b1.v, (short)0, acc1, false, false);
    acc2 = __builtin_amdgcn_wmma_f32_16x16x32_f16(false, a, false, b2.v, (short)0, acc2, false, false);
    acc3 = __builtin_amdgcn_wmma_f32_16x16x32_f16(false, a, false, b3.v, (short)0, acc3, false, false);
  }

  const int rbase = m0 + half * 8;
#pragma unroll
  for (int r = 0; r < 8; ++r) {
    float* op = out + (size_t)(rbase + r) * CHW + c * NH + col;
    op[0]  = acc0[r];
    op[16] = acc1[r];
    op[32] = acc2[r];
    op[48] = acc3[r];
  }
}

// ---------------- layernorm over H=64, one wave per (n,c) ---------------

__global__ void __launch_bounds__(256)
k_layernorm(const float* __restrict__ in, float* __restrict__ out,
            const float* __restrict__ gamma, const float* __restrict__ beta,
            int relu) {
  int lane = threadIdx.x & 31;
  int wave = threadIdx.x >> 5;
  int id = blockIdx.x * 8 + wave;           // (n * NCH + c)
  if (id >= NNODES * NCH) return;
  const float* p = in + (size_t)id * NH;
  float v0 = p[lane], v1 = p[lane + 32];
  float s = v0 + v1;
#pragma unroll
  for (int off = 16; off; off >>= 1) s += __shfl_xor(s, off, 32);
  float mu = s * (1.0f / 64.0f);
  float d0 = v0 - mu, d1 = v1 - mu;
  float q = d0 * d0 + d1 * d1;
#pragma unroll
  for (int off = 16; off; off >>= 1) q += __shfl_xor(q, off, 32);
  float inv = rsqrtf(q * (1.0f / 64.0f) + LN_EPS);
  float r0 = d0 * inv * gamma[lane] + beta[lane];
  float r1 = d1 * inv * gamma[lane + 32] + beta[lane + 32];
  if (relu) { r0 = fmaxf(r0, 0.0f); r1 = fmaxf(r1, 0.0f); }
  float* o = out + (size_t)id * NH;
  o[lane] = r0;
  o[lane + 32] = r1;
}

// ------------------------------------------------------------------------

extern "C" void kernel_launch(void* const* d_in, const int* in_sizes, int n_in,
                              void* d_out, int out_size, void* d_ws, size_t ws_size,
                              hipStream_t stream) {
  (void)in_sizes; (void)n_in; (void)out_size; (void)ws_size;
  const float*     x     = (const float*)d_in[0];
  const long long* ei    = (const long long*)d_in[1];   // int64 edge_index [2,E]
  const float*     omega = (const float*)d_in[2];
  const float*     proj  = (const float*)d_in[3];
  const float*     W1    = (const float*)d_in[4];
  const float*     W2    = (const float*)d_in[5];
  const float*     gamma = (const float*)d_in[6];
  const float*     beta  = (const float*)d_in[7];
  float* out = (float*)d_out;

  char* ws = (char*)d_ws;
  size_t off = 0;
  auto alloc = [&](size_t bytes) -> char* {
    char* p = ws + off;
    off = (off + bytes + 255) & ~(size_t)255;
    return p;
  };
  _Float16* xh    = (_Float16*)alloc((size_t)NNODES * DIN * 2);      // 25.6 MB
  _Float16* projT = (_Float16*)alloc((size_t)DIN * CHW * 2);         // 256 KB
  _Float16* W1T   = (_Float16*)alloc((size_t)NCH * NH * NH * 2);     // 64 KB
  _Float16* W2T   = (_Float16*)alloc((size_t)NCH * NH * NH * 2);     // 64 KB
  float*    hbuf  = (float*)alloc((size_t)NNODES * CHW * 4);         // 102.4 MB
  float*    accb  = out;  // reuse output buffer as scatter accumulator

  // precision / layout conversion
  k_cvt_x<<<(NNODES * DIN + 255) / 256, 256, 0, stream>>>(x, xh, NNODES * DIN);
  k_cvt_projT<<<(DIN * CHW + 255) / 256, 256, 0, stream>>>(proj, projT);
  k_cvt_WT<<<(NCH * NH * NH + 255) / 256, 256, 0, stream>>>(W1, W1T);
  k_cvt_WT<<<(NCH * NH * NH + 255) / 256, 256, 0, stream>>>(W2, W2T);

  // layer 0: projection
  k_proj_gemm<<<NNODES / 16, 256, 0, stream>>>(xh, projT, hbuf);

  // layer 0: conv -> bmm(W1) -> LN+ReLU
  hipMemsetAsync(accb, 0, (size_t)NNODES * CHW * 4, stream);
  k_scatter<<<(NEDGES * 32) / 256, 256, 0, stream>>>(hbuf, ei, omega, accb);
  k_bmm<<<NNODES / 16, 256, 0, stream>>>(accb, W1T, hbuf);
  k_layernorm<<<(NNODES * NCH) / 8, 256, 0, stream>>>(hbuf, hbuf, gamma, beta, 1);

  // layer 1: conv -> bmm(W2) -> LN
  hipMemsetAsync(accb, 0, (size_t)NNODES * CHW * 4, stream);
  k_scatter<<<(NEDGES * 32) / 256, 256, 0, stream>>>(hbuf, ei, omega, accb);
  k_bmm<<<NNODES / 16, 256, 0, stream>>>(accb, W2T, hbuf);
  k_layernorm<<<(NNODES * NCH) / 8, 256, 0, stream>>>(hbuf, out, gamma, beta, 0);
}
